// SelfAttention_25546465477228
// MI455X (gfx1250) — matrix operand order repaired and checked by reading, not compiled, not run on hardware
//
#include <hip/hip_runtime.h>
#include <hip/hip_bf16.h>
#include <math.h>

// ---------------------------------------------------------------------------
// Self-attention (softmax over query axis), CDNA5 / gfx1250, bf16 WMMA.
//   K0 : convert Wq/Wk/Wv fp32 -> bf16 (once; W reused 1024x by K1)
//   K1 : QKV projection (x[16384x1024] * W^T + b) -> bf16 Q, K, V^T in ws
//   K2 : partial per-(b,s)-column softmax stats over a q-chunk (split 4x)
//   K2R: combine partial (m, Z) -> final stats
//   K3 : partial out over an s-chunk (split 2x): P = exp(S-m)/Z, P @ V via WMMA
//   K3R: sum the two fp32 partial outputs
// ---------------------------------------------------------------------------

#define L_DIM 2048
#define B_DIM 8
#define D_DIM 1024
#define K_DIM 64
#define SCALE 0.125f      // 1/sqrt(64)
#define NEGF  (-1.0e30f)  // finite "-inf" sentinel (exp underflows to exact 0)
#define QSPLIT 4
#define SSPLIT 2
#define BL    (B_DIM * L_DIM)                 // 16384 stat entries
#define OUTN  ((size_t)L_DIM * B_DIM * K_DIM) // 1M output elements

typedef __attribute__((ext_vector_type(16))) __bf16 v16bf;
typedef __attribute__((ext_vector_type(8)))  float  v8f;

union Frag16 {                 // one WMMA 16-bit operand fragment (16 bf16 / lane)
    v16bf          v;
    unsigned short s[16];
    unsigned int   u[8];
    uint4          q[2];       // two 16-byte halves for b128 loads
};

#if defined(__has_builtin)
#  if __has_builtin(__builtin_amdgcn_cvt_pk_bf16_f32)
#    define HAVE_CVT_PK_BF16 1
#  endif
#endif

// two fp32 -> packed bf16x2 (lo in [15:0], hi in [31:16])
__device__ __forceinline__ unsigned int pack_bf16x2(float lo, float hi) {
#ifdef HAVE_CVT_PK_BF16
    auto r = __builtin_amdgcn_cvt_pk_bf16_f32(lo, hi);   // v_cvt_pk_bf16_f32 (RNE)
    return __builtin_bit_cast(unsigned int, r);
#else
    union { float f; unsigned int u; } a, b; a.f = lo; b.f = hi;
    const unsigned int ua = a.u + 0x8000u;               // round-half-up to bf16
    const unsigned int ub = b.u + 0x8000u;
    return __builtin_amdgcn_perm(ub, ua, 0x07060302u);   // v_perm_b32: hi16 of each
#endif
}

__device__ __forceinline__ unsigned short f2bf(float f) {   // scalar, round-half-up
    union { float f; unsigned int u; } c; c.f = f;
    return (unsigned short)((c.u + 0x8000u) >> 16);
}

__device__ __forceinline__ v8f wmma_bf16(const v16bf& a, const v16bf& b, const v8f& c) {
    return __builtin_amdgcn_wmma_f32_16x16x32_bf16(false, a, false, b, (short)0, c,
                                                   false, false);
}

// ---------------------------------------------------------------------------
// Kernel 0: W fp32 -> bf16. grid = (D_DIM*K_DIM/2/256, 3). One uint per thread.
// ---------------------------------------------------------------------------
__global__ __launch_bounds__(256) void convert_w_kernel(
    const float* __restrict__ Wq, const float* __restrict__ Wk,
    const float* __restrict__ Wv, unsigned int* __restrict__ Wbf)
{
    const int p   = blockIdx.y;                       // 0=Q, 1=K, 2=V
    const int tid = blockIdx.x * 256 + threadIdx.x;   // pair index
    const float* src = (p == 0) ? Wq : (p == 1) ? Wk : Wv;
    const float2 f = *(const float2*)(src + 2 * tid);
    Wbf[(size_t)p * (K_DIM * D_DIM / 2) + tid] = pack_bf16x2(f.x, f.y);
}

// ---------------------------------------------------------------------------
// Kernel 1: QKV projection. grid = (M/16, 4 col-tiles), block = 32 (1 wave).
// ---------------------------------------------------------------------------
__global__ __launch_bounds__(32) void qkv_proj_kernel(
    const float* __restrict__ x,
    const unsigned short* __restrict__ Wbf,
    const float* __restrict__ bq, const float* __restrict__ bk,
    const float* __restrict__ bv,
    unsigned short* __restrict__ Qw, unsigned short* __restrict__ Kw,
    unsigned short* __restrict__ Vtw)
{
    const int lane = threadIdx.x & 31;
    const int half = lane >> 4;
    const int n    = lane & 15;
    const int m0   = blockIdx.x * 16;           // rows of flattened (l,b)
    const int ct   = blockIdx.y;                // 16-wide k column tile
    const int arow = m0 + n;                    // A-matrix row for this lane
    const int kcol = ct * 16 + n;               // B-matrix column (output k)

    const float* bias[3] = {bq, bk, bv};

    v8f acc[3] = {};

    for (int d0 = 0; d0 < D_DIM; d0 += 32) {
        // A fragment: row arow, one 16-float contiguous segment -> 8 packed cvts
        Frag16 a;
        {
            const float* xr = x + (size_t)arow * D_DIM + d0;
            const float4 s0a = *(const float4*)(xr + half * 16);
            const float4 s0b = *(const float4*)(xr + half * 16 + 4);
            const float4 s1a = *(const float4*)(xr + half * 16 + 8);
            const float4 s1b = *(const float4*)(xr + half * 16 + 12);
            a.u[0] = pack_bf16x2(s0a.x, s0a.y);
            a.u[1] = pack_bf16x2(s0a.z, s0a.w);
            a.u[2] = pack_bf16x2(s0b.x, s0b.y);
            a.u[3] = pack_bf16x2(s0b.z, s0b.w);
            a.u[4] = pack_bf16x2(s1a.x, s1a.y);
            a.u[5] = pack_bf16x2(s1a.z, s1a.w);
            a.u[6] = pack_bf16x2(s1b.x, s1b.y);
            a.u[7] = pack_bf16x2(s1b.z, s1b.w);
        }
#pragma unroll
        for (int p = 0; p < 3; ++p) {
            // B fragment: 16 contiguous bf16 of W row kcol (pre-converted)
            Frag16 bfr;
            const unsigned short* wr =
                Wbf + (size_t)p * (K_DIM * D_DIM) + (size_t)kcol * D_DIM + d0 + half * 16;
            bfr.q[0] = *(const uint4*)(wr);
            bfr.q[1] = *(const uint4*)(wr + 8);
            acc[p] = wmma_bf16(a.v, bfr.v, acc[p]);
        }
    }

    // bias add + scatter stores (bf16). m = l*B + b  ->  l = m>>3, b = m&7.
#pragma unroll
    for (int p = 0; p < 3; ++p) {
        const float bb = bias[p][kcol];
#pragma unroll
        for (int r = 0; r < 8; ++r) {
            const int m  = m0 + r + 8 * half;
            const int l  = m >> 3;
            const int bi = m & 7;
            const unsigned short h = f2bf(acc[p][r] + bb);
            if (p == 0)      Qw [((size_t)bi * L_DIM + l) * K_DIM + kcol] = h;
            else if (p == 1) Kw [((size_t)bi * L_DIM + l) * K_DIM + kcol] = h;
            else             Vtw[((size_t)bi * K_DIM + kcol) * L_DIM + l] = h; // V^T
        }
    }
}

// ---------------------------------------------------------------------------
// Kernel 2: partial softmax stats over q (axis=1 softmax) for one q-chunk.
// One wave per (b, s-tile, q-chunk); 4096 waves. Branch-free online update.
// ---------------------------------------------------------------------------
__global__ __launch_bounds__(128) void softmax_stats_kernel(
    const unsigned short* __restrict__ Qw, const unsigned short* __restrict__ Kw,
    const unsigned char* __restrict__ mask,
    float* __restrict__ statMp, float* __restrict__ statZp)
{
    const int lane  = threadIdx.x & 31;
    const int half  = lane >> 4;
    const int n     = lane & 15;
    const int wid   = blockIdx.x * 4 + (threadIdx.x >> 5);
    const int qc    = wid & (QSPLIT - 1);
    const int stile = (wid >> 2) & 127;
    const int b     = wid >> 9;
    const int scol  = stile * 16 + n;

    const unsigned short* Qb = Qw + (size_t)b * L_DIM * K_DIM;
    const unsigned short* Kb = Kw + (size_t)b * L_DIM * K_DIM;
    const unsigned char*  mb = mask + (size_t)b * L_DIM * L_DIM;

    // B fragment of K^T depends only on scol: preload both k-steps.
    Frag16 bK[2];
#pragma unroll
    for (int kk = 0; kk < 2; ++kk) {
        const unsigned short* kr = Kb + (size_t)scol * K_DIM + kk * 32 + half * 16;
        bK[kk].q[0] = *(const uint4*)(kr);
        bK[kk].q[1] = *(const uint4*)(kr + 8);
    }

    float m = NEGF, Z = 0.f;
    const int qt0 = qc * (L_DIM / 16 / QSPLIT);          // 32 q-tiles per chunk
    const int qt1 = qt0 + (L_DIM / 16 / QSPLIT);

    for (int qt = qt0; qt < qt1; ++qt) {
        const int q0 = qt * 16;
        v8f acc = {};
#pragma unroll
        for (int kk = 0; kk < 2; ++kk) {
            Frag16 a;
            const unsigned short* qr = Qb + (size_t)(q0 + n) * K_DIM + kk * 32;
            a.q[0] = *(const uint4*)(qr + half * 16);
            a.q[1] = *(const uint4*)(qr + half * 16 + 8);
            acc = wmma_bf16(a.v, bK[kk].v, acc);
        }
        float vv[8];
        float tmax = NEGF;
#pragma unroll
        for (int r = 0; r < 8; ++r) {
            const int q = q0 + r + 8 * half;
            const bool msk = mb[(size_t)q * L_DIM + scol] != 0;
            vv[r] = msk ? NEGF : acc[r] * SCALE;
            tmax  = fmaxf(tmax, vv[r]);
        }
        const float Mn = fmaxf(m, tmax);
        float s = 0.f;
#pragma unroll
        for (int r = 0; r < 8; ++r) s += __expf(vv[r] - Mn);  // masked -> exact 0
        Z = Z * __expf(m - Mn) + s;
        m = Mn;
    }

    // combine the two lane halves (rows q%16 in 0..7 vs 8..15)
    const float m2 = __shfl_xor(m, 16, 32);
    const float Z2 = __shfl_xor(Z, 16, 32);
    const float M  = fmaxf(m, m2);
    const float Zt = Z * __expf(m - M) + Z2 * __expf(m2 - M);
    if (half == 0) {
        statMp[(size_t)qc * BL + b * L_DIM + scol] = M;
        statZp[(size_t)qc * BL + b * L_DIM + scol] = Zt;
    }
}

// ---------------------------------------------------------------------------
// Kernel 2R: combine QSPLIT partial stats. One thread per (b,s) entry.
// Garbage Z from fully-masked chunks is killed by exp(NEGF - M_real) == 0.
// ---------------------------------------------------------------------------
__global__ __launch_bounds__(256) void stats_reduce_kernel(
    const float* __restrict__ statMp, const float* __restrict__ statZp,
    float* __restrict__ statM, float* __restrict__ statZ)
{
    const int idx = blockIdx.x * 256 + threadIdx.x;     // 0 .. BL-1
    float M = NEGF;
#pragma unroll
    for (int i = 0; i < QSPLIT; ++i) M = fmaxf(M, statMp[(size_t)i * BL + idx]);
    float Z = 0.f;
#pragma unroll
    for (int i = 0; i < QSPLIT; ++i)
        Z += statZp[(size_t)i * BL + idx] * __expf(statMp[(size_t)i * BL + idx] - M);
    statM[idx] = M;
    statZ[idx] = Z;
}

// ---------------------------------------------------------------------------
// Kernel 3: partial out over one s-chunk. One wave per (b, q-tile, s-chunk);
// 2048 waves. Recompute S tiles, P = exp(S-m)/Z, stage P via LDS, P @ V WMMA.
// ---------------------------------------------------------------------------
__global__ __launch_bounds__(128) void attn_out_kernel(
    const unsigned short* __restrict__ Qw, const unsigned short* __restrict__ Kw,
    const unsigned short* __restrict__ Vtw,
    const unsigned char* __restrict__ mask,
    const float* __restrict__ statM, const float* __restrict__ statZ,
    float* __restrict__ outP)
{
    __shared__ unsigned short Pl[4][16 * 32];   // per-wave P staging (16q x 32s bf16)
    const int lane  = threadIdx.x & 31;
    const int wslot = threadIdx.x >> 5;
    const int half  = lane >> 4;
    const int n     = lane & 15;
    const int wid   = blockIdx.x * 4 + wslot;
    const int schunk = wid & (SSPLIT - 1);
    const int q0     = ((wid >> 1) & 127) * 16;
    const int b      = wid >> 8;

    const unsigned short* Qb  = Qw  + (size_t)b * L_DIM * K_DIM;
    const unsigned short* Kb  = Kw  + (size_t)b * L_DIM * K_DIM;
    const unsigned short* Vtb = Vtw + (size_t)b * K_DIM * L_DIM;
    const unsigned char*  mb  = mask + (size_t)b * L_DIM * L_DIM;
    const float* mS = statM + (size_t)b * L_DIM;
    const float* zS = statZ + (size_t)b * L_DIM;
    unsigned short* P = Pl[wslot];

    // Q A-fragments depend only on q0: preload both k-steps.
    Frag16 aQ[2];
#pragma unroll
    for (int kk = 0; kk < 2; ++kk) {
        const unsigned short* qr = Qb + (size_t)(q0 + n) * K_DIM + kk * 32;
        aQ[kk].q[0] = *(const uint4*)(qr + half * 16);
        aQ[kk].q[1] = *(const uint4*)(qr + half * 16 + 8);
    }

    v8f accO[4] = {};
    const int sc0 = schunk * (L_DIM / 32 / SSPLIT);     // 32 s-chunks of 32
    const int sc1 = sc0 + (L_DIM / 32 / SSPLIT);

    for (int sc = sc0; sc < sc1; ++sc) {
        const int sbase = sc * 32;
#pragma unroll
        for (int st = 0; st < 2; ++st) {
            const int scol = sbase + st * 16 + n;
            v8f accS = {};
#pragma unroll
            for (int kk = 0; kk < 2; ++kk) {
                Frag16 bfr;
                const unsigned short* kr = Kb + (size_t)scol * K_DIM + kk * 32 + half * 16;
                bfr.q[0] = *(const uint4*)(kr);
                bfr.q[1] = *(const uint4*)(kr + 8);
                accS = wmma_bf16(aQ[kk].v, bfr.v, accS);
            }
            const float ms = mS[scol];
            const float rz = 1.0f / zS[scol];
#pragma unroll
            for (int r = 0; r < 8; ++r) {
                const int q = q0 + r + 8 * half;
                const bool msk = mb[(size_t)q * L_DIM + scol] != 0;
                const float v = msk ? NEGF : accS[r] * SCALE;
                const float p = __expf(v - ms) * rz;    // masked -> exact 0
                P[(r + 8 * half) * 32 + st * 16 + n] = f2bf(p);
            }
        }
        // P (C layout) -> A fragment via LDS (in-order DS within the wave)
        Frag16 aP;
        {
            const unsigned short* pr = P + n * 32;
            aP.q[0] = *(const uint4*)(pr + half * 16);
            aP.q[1] = *(const uint4*)(pr + half * 16 + 8);
        }
#pragma unroll
        for (int ctile = 0; ctile < 4; ++ctile) {
            Frag16 bV;   // V^T row = output k, contiguous in s
            const unsigned short* vr =
                Vtb + (size_t)(ctile * 16 + n) * L_DIM + sbase + half * 16;
            bV.q[0] = *(const uint4*)(vr);
            bV.q[1] = *(const uint4*)(vr + 8);
            accO[ctile] = wmma_bf16(aP.v, bV.v, accO[ctile]);
        }
    }

    // partial out [schunk][L][B][K] fp32
    float* op = outP + (size_t)schunk * OUTN;
#pragma unroll
    for (int ctile = 0; ctile < 4; ++ctile)
#pragma unroll
        for (int r = 0; r < 8; ++r) {
            const int q = q0 + r + 8 * half;
            op[((size_t)q * B_DIM + b) * K_DIM + ctile * 16 + n] = accO[ctile][r];
        }
}

// ---------------------------------------------------------------------------
// Kernel 3R: out = outP[0] + outP[1], float4-vectorized.
// ---------------------------------------------------------------------------
__global__ __launch_bounds__(256) void out_reduce_kernel(
    const float* __restrict__ outP, float* __restrict__ out)
{
    const size_t i4 = (size_t)(blockIdx.x * 256 + threadIdx.x) * 4;
    const float4 a = *(const float4*)(outP + i4);
    const float4 b = *(const float4*)(outP + OUTN + i4);
    float4 r; r.x = a.x + b.x; r.y = a.y + b.y; r.z = a.z + b.z; r.w = a.w + b.w;
    *(float4*)(out + i4) = r;
}

// ---------------------------------------------------------------------------
extern "C" void kernel_launch(void* const* d_in, const int* in_sizes, int n_in,
                              void* d_out, int out_size, void* d_ws, size_t ws_size,
                              hipStream_t stream)
{
    // setup_inputs order: x, mask, Wk, bk, Wq, bq, Wv, bv
    const float*         x    = (const float*)d_in[0];
    const unsigned char* mask = (const unsigned char*)d_in[1];  // jnp bool = 1 byte
    const float*         Wk   = (const float*)d_in[2];
    const float*         bk   = (const float*)d_in[3];
    const float*         Wq   = (const float*)d_in[4];
    const float*         bq   = (const float*)d_in[5];
    const float*         Wv   = (const float*)d_in[6];
    const float*         bv   = (const float*)d_in[7];
    float*               out  = (float*)d_out;

    // workspace layout
    const size_t NQKV = (size_t)B_DIM * L_DIM * K_DIM;          // 1M bf16 elements
    unsigned short* Qw  = (unsigned short*)d_ws;
    unsigned short* Kw  = Qw + NQKV;
    unsigned short* Vtw = Kw + NQKV;
    unsigned short* Wbf = Vtw + NQKV;                           // 3*K*D bf16
    float* statMp = (float*)(Wbf + (size_t)3 * K_DIM * D_DIM);  // [QSPLIT][BL]
    float* statZp = statMp + (size_t)QSPLIT * BL;
    float* statM  = statZp + (size_t)QSPLIT * BL;               // [BL]
    float* statZ  = statM + BL;
    float* outP   = statZ + BL;                                 // [SSPLIT][OUTN]

    dim3 g0((K_DIM * D_DIM / 2) / 256, 3);                      // W -> bf16
    convert_w_kernel<<<g0, 256, 0, stream>>>(Wq, Wk, Wv, (unsigned int*)Wbf);

    dim3 g1((L_DIM * B_DIM) / 16, K_DIM / 16);                  // 1024 x 4 waves
    qkv_proj_kernel<<<g1, 32, 0, stream>>>(x, Wbf, bq, bk, bv, Qw, Kw, Vtw);

    const int w2 = B_DIM * (L_DIM / 16) * QSPLIT;               // 4096 waves
    softmax_stats_kernel<<<w2 / 4, 128, 0, stream>>>(Qw, Kw, mask, statMp, statZp);
    stats_reduce_kernel<<<BL / 256, 256, 0, stream>>>(statMp, statZp, statM, statZ);

    const int w3 = B_DIM * (L_DIM / 16) * SSPLIT;               // 2048 waves
    attn_out_kernel<<<w3 / 4, 128, 0, stream>>>(Qw, Kw, Vtw, mask, statM, statZ, outP);
    out_reduce_kernel<<<(int)(OUTN / 4 / 256), 256, 0, stream>>>(outP, out);
}
